// SRTKAN_44229573214887
// MI455X (gfx1250) — compile-verified, matrix operand order checked by reading, not run on hardware
//
#include <hip/hip_runtime.h>
#include <hip/hip_bf16.h>

typedef _Float16 v16h __attribute__((ext_vector_type(16)));
typedef _Float16 v4h  __attribute__((ext_vector_type(4)));
typedef float    v8f  __attribute__((ext_vector_type(8)));

#define TSEQ  512
#define BATCH 512
#define INF0  128
#define HID   256

union Frag16 { v16h v; _Float16 e[16]; int4 q[2]; };

// fast sigmoid / tanh on the v_rcp_f32 path (avoid IEEE-exact division chains)
__device__ __forceinline__ float sigf(float x)  {
  return __builtin_amdgcn_rcpf(1.0f + __expf(-x));
}
__device__ __forceinline__ float tanhfast(float x) {
  return 2.0f * __builtin_amdgcn_rcpf(1.0f + __expf(-2.0f * x)) - 1.0f;
}

__device__ __forceinline__ v8f wmma16(v16h a, v16h b, v8f c) {
  return __builtin_amdgcn_wmma_f32_16x16x32_f16(false, a, false, b, (short)0, c, false, false);
}

// Packed B-fragment: uniform base (SGPR) + per-lane offset -> two 16B global loads.
__device__ __forceinline__ v16h load_b_frag(const _Float16* __restrict__ ubase, int laneOff) {
  Frag16 u;
  u.q[0] = *(const int4*)(ubase + laneOff);
  u.q[1] = *(const int4*)(ubase + laneOff + 8);
  return u.v;
}

// A-fragment from LDS fp16 row-major [16][stride]: lane<16 holds row=lane,
// K = {ks..ks+7, ks+16..ks+23}, ks = 8*(lane>>4)  (ISA 16-bit A layout).
__device__ __forceinline__ v16h load_a_lds(const _Float16* __restrict__ base, int stride,
                                           int lane, int kbase) {
  int row = lane & 15;
  int ks  = (lane >> 4) << 3;
  const _Float16* p = base + row * stride + kbase + ks;
  Frag16 u;
  u.q[0] = *(const int4*)(p);
  u.q[1] = *(const int4*)(p + 16);
  return u.v;
}

// ---- weight pack: W (rows x K fp32, row-major) -> per-lane WMMA B fragments (fp16)
// B[k][n] = W[n][k].  Layout: [(tile*KC + kc)*32 + lane]*16 halves.
__global__ void pack_weights(const float* __restrict__ W, int K,
                             _Float16* __restrict__ out, int nTiles) {
  int idx = blockIdx.x * blockDim.x + threadIdx.x;
  int KC = K >> 5;
  int total = nTiles * KC * 32;
  if (idx >= total) return;
  int lane = idx & 31;
  int kc   = (idx >> 5) % KC;
  int tile = idx / (32 * KC);
  int n  = tile * 16 + (lane & 15);
  int ks = (lane >> 4) << 3;
  const float* wr = W + (size_t)n * K + (kc << 5);
  _Float16* dst = out + (size_t)idx * 16;
#pragma unroll
  for (int i = 0; i < 8; ++i) dst[i]     = (_Float16)wr[ks + i];
#pragma unroll
  for (int i = 0; i < 8; ++i) dst[8 + i] = (_Float16)wr[ks + 16 + i];
}

// ---- persistent 2-layer GRU scan: 32 blocks x 16 batch rows, 16 waves/block,
//      wave w owns hidden column tile ct = w; h0/h1/x_t staged in LDS fp16.
__global__ __launch_bounds__(512) void gru_scan(
    const float* __restrict__ x, const _Float16* __restrict__ packW,
    const float* __restrict__ bih0, const float* __restrict__ bhh0,
    const float* __restrict__ bih1, const float* __restrict__ bhh1,
    float* __restrict__ s_out) {
  __shared__ __align__(16) _Float16 h0s[16 * HID];   // 8 KB
  __shared__ __align__(16) _Float16 h1s[16 * HID];   // 8 KB
  __shared__ __align__(16) _Float16 xs[16 * INF0];   // 4 KB
  const int lane = threadIdx.x & 31;
  const int ct   = threadIdx.x >> 5;                 // 0..15 : column tile
  const int rowBase = blockIdx.x << 4;
  const int nc = lane & 15;
  const int mb = (lane >> 4) << 3;                   // row offset 0 or 8 (C/D layout)
  const int n  = ct * 16 + nc;                       // this lane's hidden column
  const int laneOff = lane * 16;                     // halves into a fragment blob

  for (int i = threadIdx.x; i < 16 * HID; i += 512) { h0s[i] = (_Float16)0.0f; h1s[i] = (_Float16)0.0f; }

  // uniform (per-wave) fragment-blob offsets within the pack, in halves
  const size_t oih0R = (size_t)(ct)      * 4 * 512;
  const size_t oih0Z = (size_t)(ct + 16) * 4 * 512;
  const size_t oih0N = (size_t)(ct + 32) * 4 * 512;
  const size_t ohh0R = (size_t)98304  + (size_t)(ct)      * 8 * 512;
  const size_t ohh0Z = (size_t)98304  + (size_t)(ct + 16) * 8 * 512;
  const size_t ohh0N = (size_t)98304  + (size_t)(ct + 32) * 8 * 512;
  const size_t oih1R = (size_t)294912 + (size_t)(ct)      * 8 * 512;
  const size_t oih1Z = (size_t)294912 + (size_t)(ct + 16) * 8 * 512;
  const size_t oih1N = (size_t)294912 + (size_t)(ct + 32) * 8 * 512;
  const size_t ohh1R = (size_t)491520 + (size_t)(ct)      * 8 * 512;
  const size_t ohh1Z = (size_t)491520 + (size_t)(ct + 16) * 8 * 512;
  const size_t ohh1N = (size_t)491520 + (size_t)(ct + 32) * 8 * 512;

  const float br0 = bih0[n] + bhh0[n];
  const float bz0 = bih0[HID + n] + bhh0[HID + n];
  const float bi0 = bih0[2 * HID + n];
  const float bh0 = bhh0[2 * HID + n];
  const float br1 = bih1[n] + bhh1[n];
  const float bz1 = bih1[HID + n] + bhh1[HID + n];
  const float bi1 = bih1[2 * HID + n];
  const float bh1 = bhh1[2 * HID + n];

  __syncthreads();

  for (int t = 0; t < TSEQ; ++t) {
    // Launder a zero *offset* (not the pointer!) each iteration: blocks LICM
    // from hoisting the 84 loop-invariant B-fragments into registers (which
    // spilled), while keeping the base a known-global pointer so the loads
    // lower to global_load_b128 (not flat_load).
    size_t zoff = 0;
    asm volatile("" : "+s"(zoff));
    const _Float16* wb = packW + zoff;

    // stage x[:,t,:] tile -> LDS fp16 (coalesced, 4 floats/thread)
    {
      int idx4 = threadIdx.x * 4;                   // 0..2044, covers 16*128
      int row = idx4 >> 7, k = idx4 & 127;
      const float* p = x + ((size_t)(rowBase + row) * TSEQ + t) * INF0 + k;
      float4 f = *(const float4*)p;
      v4h h; h.x = (_Float16)f.x; h.y = (_Float16)f.y; h.z = (_Float16)f.z; h.w = (_Float16)f.w;
      *(v4h*)(xs + idx4) = h;
      if (t + 1 < TSEQ) __builtin_prefetch(p + INF0, 0, 0);
    }
    __syncthreads();

    // ================= GRU layer 0 =================
    v8f aR = {}, aZ = {}, aI = {}, aH = {};
#pragma unroll 1
    for (int kc = 0; kc < 4; ++kc) {               // x_t @ Wih0^T, K=128
      v16h a = load_a_lds(xs, INF0, lane, kc * 32);
      size_t ko = (size_t)kc * 512;
      aR = wmma16(a, load_b_frag(wb + oih0R + ko, laneOff), aR);
      aZ = wmma16(a, load_b_frag(wb + oih0Z + ko, laneOff), aZ);
      aI = wmma16(a, load_b_frag(wb + oih0N + ko, laneOff), aI);
    }
#pragma unroll 1
    for (int kc = 0; kc < 8; ++kc) {               // h0 @ Whh0^T, K=256
      v16h a = load_a_lds(h0s, HID, lane, kc * 32);
      size_t ko = (size_t)kc * 512;
      aR = wmma16(a, load_b_frag(wb + ohh0R + ko, laneOff), aR);
      aZ = wmma16(a, load_b_frag(wb + ohh0Z + ko, laneOff), aZ);
      aH = wmma16(a, load_b_frag(wb + ohh0N + ko, laneOff), aH);
    }
    float nh0[8];
#pragma unroll
    for (int v = 0; v < 8; ++v) {
      float hp = (float)h0s[(v + mb) * HID + n];
      float r  = sigf(aR[v] + br0);
      float z  = sigf(aZ[v] + bz0);
      float nn = tanhfast(aI[v] + bi0 + r * (aH[v] + bh0));
      nh0[v] = (1.0f - z) * nn + z * hp;
    }
    __syncthreads();
#pragma unroll
    for (int v = 0; v < 8; ++v) h0s[(v + mb) * HID + n] = (_Float16)nh0[v];
    __syncthreads();

    // ================= GRU layer 1 =================
    aR = {}; aZ = {}; aI = {}; aH = {};
#pragma unroll 1
    for (int kc = 0; kc < 8; ++kc) {               // h0' @ Wih1^T and h1 @ Whh1^T
      size_t ko = (size_t)kc * 512;
      v16h a0 = load_a_lds(h0s, HID, lane, kc * 32);
      aR = wmma16(a0, load_b_frag(wb + oih1R + ko, laneOff), aR);
      aZ = wmma16(a0, load_b_frag(wb + oih1Z + ko, laneOff), aZ);
      aI = wmma16(a0, load_b_frag(wb + oih1N + ko, laneOff), aI);
      v16h a1 = load_a_lds(h1s, HID, lane, kc * 32);
      aR = wmma16(a1, load_b_frag(wb + ohh1R + ko, laneOff), aR);
      aZ = wmma16(a1, load_b_frag(wb + ohh1Z + ko, laneOff), aZ);
      aH = wmma16(a1, load_b_frag(wb + ohh1N + ko, laneOff), aH);
    }
    float nh1[8];
#pragma unroll
    for (int v = 0; v < 8; ++v) {
      float hp = (float)h1s[(v + mb) * HID + n];
      float r  = sigf(aR[v] + br1);
      float z  = sigf(aZ[v] + bz1);
      float nn = tanhfast(aI[v] + bi1 + r * (aH[v] + bh1));
      nh1[v] = (1.0f - z) * nn + z * hp;
    }
    __syncthreads();
#pragma unroll
    for (int v = 0; v < 8; ++v) h1s[(v + mb) * HID + n] = (_Float16)nh1[v];
    __syncthreads();
  }

  for (int i = threadIdx.x; i < 16 * HID; i += 512) {
    int r = i >> 8, c = i & 255;
    s_out[(size_t)(rowBase + r) * HID + c] = (float)h1s[i];
  }
}

// ---- dual-direction single-step LSTM (h0=c0=0 => forget gate irrelevant)
__global__ __launch_bounds__(256) void lstm_dual(
    const float* __restrict__ in,
    const float* __restrict__ wA, const float* __restrict__ biA, const float* __restrict__ bhA,
    const float* __restrict__ wB, const float* __restrict__ biB, const float* __restrict__ bhB,
    float* __restrict__ out) {
  int b = blockIdx.x;
  int j = threadIdx.x;                  // 0..255 ; first 128 = dir A, last 128 = dir B
  const float *w, *bi, *bh;
  int u;
  if (j < 128) { w = wA; bi = biA; bh = bhA; u = j; }
  else         { w = wB; bi = biB; bh = bhB; u = j - 128; }
  float gi = bi[u] + bh[u];
  float gg = bi[256 + u] + bh[256 + u];
  float go = bi[384 + u] + bh[384 + u];
  const float* xr = in + (size_t)b * 256;
  const float* wi = w + (size_t)u * 256;
  const float* wg = w + (size_t)(256 + u) * 256;
  const float* wo = w + (size_t)(384 + u) * 256;
  for (int k = 0; k < 256; k += 4) {
    float4 xv = *(const float4*)(xr + k);
    float4 a  = *(const float4*)(wi + k);
    float4 c  = *(const float4*)(wg + k);
    float4 d  = *(const float4*)(wo + k);
    gi += xv.x * a.x + xv.y * a.y + xv.z * a.z + xv.w * a.w;
    gg += xv.x * c.x + xv.y * c.y + xv.z * c.z + xv.w * c.w;
    go += xv.x * d.x + xv.y * d.y + xv.z * d.z + xv.w * d.w;
  }
  float c = sigf(gi) * tanhfast(gg);
  out[(size_t)b * 256 + j] = sigf(go) * tanhfast(c);
}

// ---- KAN linear: base = silu(x) @ Wb^T ; spline = einsum(bspl(x), Ws*scaler)
template <int INFEAT, int OUTFEAT>
__global__ void kan_layer(const float* __restrict__ x,
                          const float* __restrict__ base_w,
                          const float* __restrict__ spline_w,
                          const float* __restrict__ scaler,
                          float* __restrict__ out) {
  __shared__ float silu_s[INFEAT];
  __shared__ float bs[INFEAT][8];
  int b = blockIdx.x;
  for (int i = threadIdx.x; i < INFEAT; i += blockDim.x) {
    float xv = x[(size_t)b * INFEAT + i];
    silu_s[i] = xv * sigf(xv);
    float bb[11];
#pragma unroll
    for (int j = 0; j < 11; ++j) {
      float p0 = (float)(j - 3) * 0.4f - 1.0f;
      float p1 = (float)(j - 2) * 0.4f - 1.0f;
      bb[j] = (xv >= p0 && xv < p1) ? 1.0f : 0.0f;
    }
#pragma unroll
    for (int k = 1; k <= 3; ++k) {
#pragma unroll
      for (int j = 0; j < 11; ++j) {
        if (j < 11 - k) {
          float pj   = (float)(j - 3) * 0.4f - 1.0f;
          float pjk  = (float)(j + k - 3) * 0.4f - 1.0f;
          float pj1  = (float)(j - 2) * 0.4f - 1.0f;
          float pjk1 = (float)(j + k - 2) * 0.4f - 1.0f;
          float invA = __builtin_amdgcn_rcpf(pjk - pj);
          float invB = __builtin_amdgcn_rcpf(pjk1 - pj1);
          bb[j] = (xv - pj) * invA * bb[j] + (pjk1 - xv) * invB * bb[j + 1];
        }
      }
    }
#pragma unroll
    for (int k = 0; k < 8; ++k) bs[i][k] = bb[k];
  }
  __syncthreads();
  for (int o = threadIdx.x; o < OUTFEAT; o += blockDim.x) {
    float acc = 0.0f;
    for (int i = 0; i < INFEAT; ++i) {
      acc += silu_s[i] * base_w[(size_t)o * INFEAT + i];
      const float* sw = spline_w + ((size_t)o * INFEAT + i) * 8;
      float sp = 0.0f;
#pragma unroll
      for (int k = 0; k < 8; ++k) sp += bs[i][k] * sw[k];
      acc += scaler[(size_t)o * INFEAT + i] * sp;
    }
    out[(size_t)b * OUTFEAT + o] = acc;
  }
}

extern "C" void kernel_launch(void* const* d_in, const int* in_sizes, int n_in,
                              void* d_out, int out_size, void* d_ws, size_t ws_size,
                              hipStream_t stream) {
  (void)in_sizes; (void)n_in; (void)out_size; (void)ws_size;
  const float* x        = (const float*)d_in[0];
  const float* g0_wih   = (const float*)d_in[1];
  const float* g0_whh   = (const float*)d_in[2];
  const float* g0_bih   = (const float*)d_in[3];
  const float* g0_bhh   = (const float*)d_in[4];
  const float* g1_wih   = (const float*)d_in[5];
  const float* g1_whh   = (const float*)d_in[6];
  const float* g1_bih   = (const float*)d_in[7];
  const float* g1_bhh   = (const float*)d_in[8];
  const float* lw_ih0   = (const float*)d_in[9];
  const float* lb_ih0   = (const float*)d_in[11];
  const float* lb_hh0   = (const float*)d_in[12];
  const float* lw_ih0r  = (const float*)d_in[13];
  const float* lb_ih0r  = (const float*)d_in[15];
  const float* lb_hh0r  = (const float*)d_in[16];
  const float* lw_ih1   = (const float*)d_in[17];
  const float* lb_ih1   = (const float*)d_in[19];
  const float* lb_hh1   = (const float*)d_in[20];
  const float* lw_ih1r  = (const float*)d_in[21];
  const float* lb_ih1r  = (const float*)d_in[23];
  const float* lb_hh1r  = (const float*)d_in[24];
  const float* k1_base  = (const float*)d_in[25];
  const float* k1_spl   = (const float*)d_in[26];
  const float* k1_scal  = (const float*)d_in[27];
  const float* k2_base  = (const float*)d_in[28];
  const float* k2_spl   = (const float*)d_in[29];
  const float* k2_scal  = (const float*)d_in[30];

  _Float16* packW = (_Float16*)d_ws;                 // 688128 halves = 1376256 B
  float* wsf = (float*)((char*)d_ws + 1376256);
  float* s  = wsf;                                   // 512*256
  float* o0 = s  + (size_t)BATCH * HID;              // 512*256
  float* o1 = o0 + (size_t)BATCH * HID;              // 512*256
  float* zz = o1 + (size_t)BATCH * HID;              // 512*64

  // pack GRU weights into WMMA B fragments (fp16)
  pack_weights<<<(48 * 4 * 32 + 255) / 256, 256, 0, stream>>>(g0_wih, 128, packW, 48);
  pack_weights<<<(48 * 8 * 32 + 255) / 256, 256, 0, stream>>>(g0_whh, 256, packW + 98304, 48);
  pack_weights<<<(48 * 8 * 32 + 255) / 256, 256, 0, stream>>>(g1_wih, 256, packW + 294912, 48);
  pack_weights<<<(48 * 8 * 32 + 255) / 256, 256, 0, stream>>>(g1_whh, 256, packW + 491520, 48);

  // persistent 2-layer GRU scan over all 512 timesteps
  gru_scan<<<BATCH / 16, 512, 0, stream>>>(x, packW, g0_bih, g0_bhh, g1_bih, g1_bhh, s);

  // LSTM heads (single step, zero initial state)
  lstm_dual<<<BATCH, 256, 0, stream>>>(s,  lw_ih0, lb_ih0, lb_hh0, lw_ih0r, lb_ih0r, lb_hh0r, o0);
  lstm_dual<<<BATCH, 256, 0, stream>>>(o0, lw_ih1, lb_ih1, lb_hh1, lw_ih1r, lb_ih1r, lb_hh1r, o1);

  // KAN layers
  kan_layer<256, 64><<<BATCH, 256, 0, stream>>>(o1, k1_base, k1_spl, k1_scal, zz);
  kan_layer<64, 10><<<BATCH, 64, 0, stream>>>(zz, k2_base, k2_spl, k2_scal, (float*)d_out);
}
